// MSA_16338055594302
// MI455X (gfx1250) — compile-verified, matrix operand order checked by reading, not run on hardware
//
#include <hip/hip_runtime.h>
#include <hip/hip_bf16.h>

typedef __bf16 bf16_t;
typedef __attribute__((ext_vector_type(16))) __bf16 v16bf;
typedef __attribute__((ext_vector_type(8)))  float  v8f;

#define BATCH     8
#define SEQ       1024
#define D_MODEL   768
#define NUM_HEADS 12
#define HEAD_DIM  64
#define INV_SCALE 0.03608439182435161f   // 1/sqrt(768)

static __device__ __forceinline__ v8f wmma_bf16(v16bf a, v16bf b, v8f c) {
    return __builtin_amdgcn_wmma_f32_16x16x32_bf16(false, a, false, b,
                                                   (short)0, c, false, false);
}

// ---------------- Stage 0: fp32 -> bf16 conversion ----------------
__global__ __launch_bounds__(256) void cvt_bf16_kernel(const float* __restrict__ src,
                                                       bf16_t* __restrict__ dst, int n) {
    int i = blockIdx.x * 256 + threadIdx.x;
    if (i < n) dst[i] = (bf16_t)src[i];
}

// ---------------- Stage 1: fused QKV projection (y = x @ W^T + b) ----------------
// grid: (8192/64, 768/128, 3), block 256 (8 waves).
// Wave computes a 32x32 tile (2x2 register blocking: each fragment feeds 2 WMMAs).
__global__ __launch_bounds__(256) void qkv_kernel(
    const bf16_t* __restrict__ xb,    // [8192, 768]
    const bf16_t* __restrict__ wb,    // [3, 768, 768]  (Wq,Wk,Wv row-major)
    const float*  __restrict__ bq,
    const float*  __restrict__ bk,
    const float*  __restrict__ bv,
    bf16_t* __restrict__ qb,          // [B,H,S,64]
    bf16_t* __restrict__ kb,          // [B,H,S,64]
    bf16_t* __restrict__ vtb)         // [B,H,64,S]
{
    const int lane = threadIdx.x & 31;
    const int wave = threadIdx.x >> 5;
    const int proj = blockIdx.z;                          // 0=q 1=k 2=v
    const int mBase = blockIdx.x * 64  + (wave >> 2) * 32;
    const int nBase = blockIdx.y * 128 + (wave & 3) * 32;
    const int kOff  = (lane >> 4) * 16;

    const bf16_t* Arow0 = xb + (size_t)(mBase + (lane & 15)) * D_MODEL + kOff;
    const bf16_t* Arow1 = Arow0 + (size_t)16 * D_MODEL;
    const bf16_t* W = wb + (size_t)proj * D_MODEL * D_MODEL;
    const bf16_t* Brow0 = W + (size_t)(nBase + (lane & 15)) * D_MODEL + kOff;
    const bf16_t* Brow1 = Brow0 + (size_t)16 * D_MODEL;

    v8f acc00 = {}, acc01 = {}, acc10 = {}, acc11 = {};
    for (int k = 0; k < D_MODEL; k += 32) {
        v16bf a0 = *(const v16bf*)(Arow0 + k);
        v16bf a1 = *(const v16bf*)(Arow1 + k);
        v16bf b0 = *(const v16bf*)(Brow0 + k);
        v16bf b1 = *(const v16bf*)(Brow1 + k);
        acc00 = wmma_bf16(a0, b0, acc00);
        acc01 = wmma_bf16(a0, b1, acc01);
        acc10 = wmma_bf16(a1, b0, acc10);
        acc11 = wmma_bf16(a1, b1, acc11);
    }

    const float* bias = (proj == 0) ? bq : (proj == 1) ? bk : bv;
    const int mHalf = (lane >> 4) * 8;

    for (int mi = 0; mi < 2; ++mi) {
        for (int ni = 0; ni < 2; ++ni) {
            const v8f acc = (mi == 0) ? (ni == 0 ? acc00 : acc01)
                                      : (ni == 0 ? acc10 : acc11);
            const int n  = nBase + ni * 16 + (lane & 15);  // fixed per lane
            const int h  = n >> 6;
            const int hd = n & 63;
            const float bn = bias[n];
            if (proj == 2) {
                // V transposed: [B,H,64,S]
                for (int r = 0; r < 8; ++r) {
                    int m = mBase + mi * 16 + mHalf + r;
                    int bidx = m >> 10, s = m & 1023;
                    vtb[(((size_t)bidx * NUM_HEADS + h) * HEAD_DIM + hd) * SEQ + s] =
                        (bf16_t)(acc[r] + bn);
                }
            } else {
                bf16_t* dst = (proj == 0) ? qb : kb;       // [B,H,S,64]
                for (int r = 0; r < 8; ++r) {
                    int m = mBase + mi * 16 + mHalf + r;
                    int bidx = m >> 10, s = m & 1023;
                    dst[(((size_t)bidx * NUM_HEADS + h) * SEQ + s) * HEAD_DIM + hd] =
                        (bf16_t)(acc[r] + bn);
                }
            }
        }
    }
}

// ---------------- Stage 2: attention ----------------
// grid: (S/32, H, B), block 256 (8 waves). 64KB static LDS of bf16 scores.
__global__ __launch_bounds__(256) void attn_kernel(
    const bf16_t* __restrict__ qb,
    const bf16_t* __restrict__ kb,
    const bf16_t* __restrict__ vtb,
    float* __restrict__ outp,         // [B,S,768]
    float* __restrict__ Aout)         // [B,H,S,S]
{
    __shared__ bf16_t sc[32][SEQ];    // exactly 64 KB

    const int lane = threadIdx.x & 31;
    const int wave = threadIdx.x >> 5;
    const int h = blockIdx.y;
    const int b = blockIdx.z;
    const size_t bh = (size_t)b * NUM_HEADS + h;
    const int sBase = blockIdx.x * 32;
    const int kOff = (lane >> 4) * 16;

    const bf16_t* Q  = qb  + bh * SEQ * HEAD_DIM;
    const bf16_t* K  = kb  + bh * SEQ * HEAD_DIM;
    const bf16_t* Vt = vtb + bh * HEAD_DIM * SEQ;

    // ---- Phase 1: scores = (Q_tile @ K^T) * inv_scale -> LDS (bf16) ----
    // Wave strip: fixed mTile (Q fragments loaded ONCE), 16 K-column tiles.
    {
        const int mTile  = wave & 1;            // 0..1
        const int nGroup = wave >> 1;           // 0..3 -> nTiles nGroup*16..+16
        const int mRow = sBase + mTile * 16 + (lane & 15);
        const v16bf a0 = *(const v16bf*)(Q + (size_t)mRow * HEAD_DIM + kOff);
        const v16bf a1 = *(const v16bf*)(Q + (size_t)mRow * HEAD_DIM + 32 + kOff);
        const int mLoc = mTile * 16 + (lane >> 4) * 8;

        for (int j = 0; j < 16; ++j) {
            const int nCol = (nGroup * 16 + j) * 16 + (lane & 15);
            v16bf b0 = *(const v16bf*)(K + (size_t)nCol * HEAD_DIM + kOff);
            v16bf b1 = *(const v16bf*)(K + (size_t)nCol * HEAD_DIM + 32 + kOff);
            v8f acc = {};
            acc = wmma_bf16(a0, b0, acc);
            acc = wmma_bf16(a1, b1, acc);
            for (int r = 0; r < 8; ++r)
                sc[mLoc + r][nCol] = (bf16_t)(acc[r] * INV_SCALE);
        }
    }
    __syncthreads();

    // ---- Phase 2+3: per-row softmax; write normalized A (fp32, coalesced,
    //      non-temporal); keep normalized bf16 copy in LDS. 4 rows per wave. ----
    for (int rr = 0; rr < 4; ++rr) {
        const int row = wave * 4 + rr;
        float mx = -3.402823e38f;
        for (int i = 0; i < 32; ++i)
            mx = fmaxf(mx, (float)sc[row][lane + 32 * i]);
        for (int off = 16; off > 0; off >>= 1)
            mx = fmaxf(mx, __shfl_xor(mx, off, 32));
        float sum = 0.0f;
        for (int i = 0; i < 32; ++i)
            sum += __expf((float)sc[row][lane + 32 * i] - mx);
        for (int off = 16; off > 0; off >>= 1)
            sum += __shfl_xor(sum, off, 32);
        const float rcp = 1.0f / sum;

        float* Ar = Aout + (bh * SEQ + (size_t)(sBase + row)) * SEQ;
        for (int i = 0; i < 32; ++i) {
            const int col = lane + 32 * i;
            float a = __expf((float)sc[row][col] - mx) * rcp;
            __builtin_nontemporal_store(a, Ar + col);   // streaming, keep L2 for Q/K/Vt
            sc[row][col] = (bf16_t)a;
        }
    }
    __syncthreads();

    // ---- Phase 4: out_tile = A_tile @ V  (32x64), one 16x16 tile per wave ----
    {
        const int mRow = (wave >> 2) * 16 + (lane & 15);   // LDS row
        const int nCol = (wave & 3) * 16 + (lane & 15);    // head-dim index
        v8f acc = {};
        for (int t = 0; t < SEQ; t += 32) {
            v16bf a  = *(const v16bf*)(&sc[mRow][t + kOff]);
            v16bf bb = *(const v16bf*)(Vt + (size_t)nCol * SEQ + t + kOff);
            acc = wmma_bf16(a, bb, acc);
        }
        const int m0 = (wave >> 2) * 16 + (lane >> 4) * 8;
        for (int r = 0; r < 8; ++r) {
            const int s = sBase + m0 + r;
            __builtin_nontemporal_store(
                acc[r], outp + ((size_t)b * SEQ + s) * D_MODEL + h * HEAD_DIM + nCol);
        }
    }
}

extern "C" void kernel_launch(void* const* d_in, const int* in_sizes, int n_in,
                              void* d_out, int out_size, void* d_ws, size_t ws_size,
                              hipStream_t stream) {
    const float* x  = (const float*)d_in[0];
    const float* Wq = (const float*)d_in[1];
    const float* bq = (const float*)d_in[2];
    const float* Wk = (const float*)d_in[3];
    const float* bk = (const float*)d_in[4];
    const float* Wv = (const float*)d_in[5];
    const float* bv = (const float*)d_in[6];

    const int nx = BATCH * SEQ * D_MODEL;      // 6,291,456
    const int nw = D_MODEL * D_MODEL;          // 589,824

    char* ws = (char*)d_ws;
    bf16_t* xb  = (bf16_t*)(ws);                                  // 12,582,912 B
    bf16_t* wb  = (bf16_t*)(ws + (size_t)nx * 2);                 //  3,538,944 B
    bf16_t* qb  = (bf16_t*)(ws + (size_t)nx * 2 + (size_t)nw * 6);
    bf16_t* kb  = qb + (size_t)nx;
    bf16_t* vtb = kb + (size_t)nx;                                // total ~54 MB

    cvt_bf16_kernel<<<nx / 256, 256, 0, stream>>>(x,  xb, nx);
    cvt_bf16_kernel<<<nw / 256, 256, 0, stream>>>(Wq, wb,          nw);
    cvt_bf16_kernel<<<nw / 256, 256, 0, stream>>>(Wk, wb + nw,     nw);
    cvt_bf16_kernel<<<nw / 256, 256, 0, stream>>>(Wv, wb + 2 * nw, nw);

    qkv_kernel<<<dim3((BATCH * SEQ) / 64, D_MODEL / 128, 3), 256, 0, stream>>>(
        xb, wb, bq, bk, bv, qb, kb, vtb);

    float* outp = (float*)d_out;
    float* Aout = outp + (size_t)BATCH * SEQ * D_MODEL;
    attn_kernel<<<dim3(SEQ / 32, NUM_HEADS, BATCH), 256, 0, stream>>>(
        qb, kb, vtb, outp, Aout);
}
